// Attention_29016799052325
// MI455X (gfx1250) — matrix operand order repaired and checked by reading, not compiled, not run on hardware
//
#include <hip/hip_runtime.h>

// ---------------------------------------------------------------------------
// Types for CDNA5 WMMA (wave32): V_WMMA_F32_16X16X32_BF16
// ---------------------------------------------------------------------------
typedef __bf16 bf16_t;
typedef __attribute__((ext_vector_type(8)))  bf16_t v8bf;
typedef __attribute__((ext_vector_type(16))) bf16_t v16bf;
typedef __attribute__((ext_vector_type(8)))  float  v8f;
typedef unsigned int v4u __attribute__((ext_vector_type(4)));
typedef int          v8i __attribute__((ext_vector_type(8)));
typedef int          v4i __attribute__((ext_vector_type(4)));

union Frag16 { v16bf v; v8bf h[2]; unsigned short us[16]; };

__device__ inline unsigned short bfbits(float f) {
  union { float f; unsigned u; } x; x.f = f;
  unsigned u = x.u + 0x7FFFu + ((x.u >> 16) & 1u);   // round-to-nearest-even
  return (unsigned short)(u >> 16);
}
__device__ inline bf16_t f2bf(float f) {
  union { unsigned short s; bf16_t b; } y; y.s = bfbits(f); return y.b;
}
__device__ inline unsigned pk2(float a, float b) {
  return (unsigned)bfbits(a) | ((unsigned)bfbits(b) << 16);
}
__device__ inline v8f vzero8() {
  v8f z = {0.f,0.f,0.f,0.f,0.f,0.f,0.f,0.f}; return z;
}

// Problem constants
#define BATCH 4
#define SEQ   2048
#define DMODEL 1024
#define NHEAD 16
#define DHEAD 64
#define QSZ   (BATCH*SEQ*DMODEL)     // 8,388,608 elements per [B,H,S,dh] block

// ---------------------------------------------------------------------------
// Tensor Data Mover: one-instruction 2D bf16 tile copy global->LDS with row
// padding (CDNA5 ISA D# layout).
//   pad_interval code p: pad every 2^(p+1) DWORDs; pad_amount code a: a+1 DWORDs
// ---------------------------------------------------------------------------
__device__ inline void tdm_load_2d(unsigned lds_addr, const void* gaddr,
                                   unsigned tensor_d0, unsigned tensor_d1,
                                   unsigned tile_d0, unsigned tile_d1,
                                   unsigned d0_stride,
                                   unsigned pad_interval_code,
                                   unsigned pad_amount_code)
{
  unsigned long long ga = (unsigned long long)(uintptr_t)gaddr;
  v4u g0;
  g0.x = 1u;                                        // count=1, user desc
  g0.y = lds_addr;                                  // LDS byte address
  g0.z = (unsigned)(ga & 0xFFFFFFFFu);              // global_addr[31:0]
  g0.w = (unsigned)((ga >> 32) & 0x01FFFFFFu) | (2u << 30);  // [56:32] | type=2
  v8i g1;
  g1[0] = (int)((1u << 16)                          // data_size = 2 bytes
              | (1u << 20)                          // pad_enable
              | (pad_interval_code << 22)
              | (pad_amount_code << 25));
  g1[1] = (int)((tensor_d0 & 0xFFFFu) << 16);                       // dim0[15:0]
  g1[2] = (int)(((tensor_d0 >> 16) & 0xFFFFu) |
                ((tensor_d1 & 0xFFFFu) << 16));                     // dim0 hi | dim1 lo
  g1[3] = (int)(((tensor_d1 >> 16) & 0xFFFFu) |
                ((tile_d0 & 0xFFFFu) << 16));                       // dim1 hi | tile0
  g1[4] = (int)(tile_d1 & 0xFFFFu);                                 // tile1 | tile2=0
  g1[5] = (int)d0_stride;                                           // dim0_stride lo32
  g1[6] = 0;                                                        // stride hi | dim1_stride lo
  g1[7] = 0;
  v4i z4 = {0,0,0,0};
#if __clang_major__ >= 23
  v8i z8 = {0,0,0,0,0,0,0,0};
  __builtin_amdgcn_tensor_load_to_lds(g0, g1, z4, z4, z8, 0);
#else
  __builtin_amdgcn_tensor_load_to_lds(g0, g1, z4, z4, 0);
#endif
}

// ---------------------------------------------------------------------------
// Pre-pass 1: f32 -> bf16 elementwise (x)
// ---------------------------------------------------------------------------
__global__ __launch_bounds__(256) void cvt_bf16_kernel(
    const float* __restrict__ in, bf16_t* __restrict__ out, int n4)
{
  int i = blockIdx.x * blockDim.x + threadIdx.x;
  if (i < n4) {
    float4 v = ((const float4*)in)[i];
    uint2 p; p.x = pk2(v.x, v.y); p.y = pk2(v.z, v.w);
    ((uint2*)out)[i] = p;
  }
}

// ---------------------------------------------------------------------------
// Pre-pass 2: w[K,N] f32 -> wT[N,K] bf16 (tiled transpose via LDS)
// ---------------------------------------------------------------------------
__global__ __launch_bounds__(256) void cvt_transpose_kernel(
    const float* __restrict__ w, bf16_t* __restrict__ wT, int K, int N)
{
  __shared__ float tile[32][33];
  int k0 = blockIdx.y * 32, n0 = blockIdx.x * 32;
  int tx = threadIdx.x & 31, ty = threadIdx.x >> 5;   // ty: 0..7
  #pragma unroll
  for (int r = ty; r < 32; r += 8)
    tile[r][tx] = w[(size_t)(k0 + r) * N + n0 + tx];
  __syncthreads();
  #pragma unroll
  for (int r = ty; r < 32; r += 8)
    wT[(size_t)(n0 + r) * K + k0 + tx] = f2bf(tile[tx][r]);
}

// ---------------------------------------------------------------------------
// bf16-WMMA GEMM: C[M,N] = A[M,K] * BT[N,K]^T + bias.
// Double-buffered TDM staging: DMA of tile i+1 overlaps WMMA of tile i,
// synchronized on TENSORcnt (in-order per wave: wait<=2 => prev pair landed).
// Block: 256 threads (8 waves), tile 128x128, K-step 32.
// mode 0 (QKV): Q -> qb(bf16 [B,H,S,dh]); K -> present0(f32) + kb(bf16);
//               V -> present1(f32) + vt(bf16 [B,H,dh,S] transposed)
// mode 1: plain row-major f32 out
// ---------------------------------------------------------------------------
#define LSTR 40   // LDS row stride (bf16): 64B row + 16B TDM pad = 80B

__global__ __launch_bounds__(256) void gemm_bf16_kernel(
    const bf16_t* __restrict__ A, const bf16_t* __restrict__ BT,
    const float* __restrict__ bias,
    float* __restrict__ outF, bf16_t* __restrict__ qb,
    float* __restrict__ outK, bf16_t* __restrict__ kb,
    float* __restrict__ outV, bf16_t* __restrict__ vt,
    int Mdim, int Ndim, int Kdim, int mode)
{
  __shared__ bf16_t Ab[2][128 * LSTR];   // [m][k], double buffered
  __shared__ bf16_t Bb[2][128 * LSTR];   // [n][k], double buffered

  const int tid  = threadIdx.x;
  const int lane = tid & 31;
  const int wid  = tid >> 5;
  const int m0 = blockIdx.y * 128;
  const int n0 = blockIdx.x * 128;
  const int wm = wid >> 2;            // 0..1 -> 64 rows
  const int wn = wid & 3;             // 0..3 -> 32 cols
  const int ml = lane & 15;
  const int abase = (lane < 16) ? 0 : 8;   // A-frag K base
  const int kbb   = (lane < 16) ? 0 : 16;  // unused (B-frag loads use abase)
  const unsigned ldsA[2] = { (unsigned)(uintptr_t)&Ab[0][0],
                             (unsigned)(uintptr_t)&Ab[1][0] };
  const unsigned ldsB[2] = { (unsigned)(uintptr_t)&Bb[0][0],
                             (unsigned)(uintptr_t)&Bb[1][0] };

  v8f acc[4][2];
  #pragma unroll
  for (int mt = 0; mt < 4; ++mt)
    #pragma unroll
    for (int nt = 0; nt < 2; ++nt) acc[mt][nt] = vzero8();

  const int nk = Kdim >> 5;
  if (wid == 0) {   // prologue: stage tile 0; TDM is wave-level, EXEC ignored
    tdm_load_2d(ldsA[0], A + (size_t)m0 * Kdim,
                (unsigned)Kdim, (unsigned)Mdim, 32u, 128u, (unsigned)Kdim, 3u, 3u);
    tdm_load_2d(ldsB[0], BT + (size_t)n0 * Kdim,
                (unsigned)Kdim, (unsigned)Ndim, 32u, 128u, (unsigned)Kdim, 3u, 3u);
  }

  for (int i = 0; i < nk; ++i) {
    const int cur = i & 1;
    if (wid == 0) {
      if (i + 1 < nk) {   // prefetch tile i+1 into the other buffer
        int k1 = (i + 1) << 5;
        tdm_load_2d(ldsA[cur ^ 1], A + (size_t)m0 * Kdim + k1,
                    (unsigned)Kdim, (unsigned)Mdim, 32u, 128u, (unsigned)Kdim, 3u, 3u);
        tdm_load_2d(ldsB[cur ^ 1], BT + (size_t)n0 * Kdim + k1,
                    (unsigned)Kdim, (unsigned)Ndim, 32u, 128u, (unsigned)Kdim, 3u, 3u);
        __builtin_amdgcn_s_wait_tensorcnt(2);   // tile i landed; i+1 in flight
      } else {
        __builtin_amdgcn_s_wait_tensorcnt(0);
      }
    }
    __syncthreads();

    const bf16_t* Abuf = &Ab[cur][0];
    const bf16_t* Bbuf = &Bb[cur][0];
    Frag16 af[4], bfr[2];
    #pragma unroll
    for (int mt = 0; mt < 4; ++mt) {
      int row = wm*64 + mt*16 + ml;
      af[mt].h[0] = *(const v8bf*)&Abuf[row*LSTR + abase];
      af[mt].h[1] = *(const v8bf*)&Abuf[row*LSTR + 16 + abase];
    }
    #pragma unroll
    for (int nt = 0; nt < 2; ++nt) {
      int n = wn*32 + nt*16 + ml;
      bfr[nt].h[0] = *(const v8bf*)&Bbuf[n*LSTR + abase];
      bfr[nt].h[1] = *(const v8bf*)&Bbuf[n*LSTR + 16 + abase];
    }
    #pragma unroll
    for (int mt = 0; mt < 4; ++mt)
      #pragma unroll
      for (int nt = 0; nt < 2; ++nt)
        acc[mt][nt] = __builtin_amdgcn_wmma_f32_16x16x32_bf16(
            false, af[mt].v, false, bfr[nt].v, (short)0, acc[mt][nt],
            false, false);
    __syncthreads();
  }

  // Epilogue: C layout -> VGPR r holds row r + (lane<16?0:8), col = lane%16
  const int rofs = (lane < 16) ? 0 : 8;
  #pragma unroll
  for (int mt = 0; mt < 4; ++mt) {
    #pragma unroll
    for (int nt = 0; nt < 2; ++nt) {
      int cg = n0 + wn*32 + nt*16 + ml;
      float bv = bias[cg];
      #pragma unroll
      for (int r = 0; r < 8; ++r) {
        int rg = m0 + wm*64 + mt*16 + r + rofs;
        float v = acc[mt][nt][r] + bv;
        if (mode == 1) {
          outF[(size_t)rg * Ndim + cg] = v;
        } else {
          int b = rg >> 11, s = rg & 2047;
          if (cg < DMODEL) {
            int h = cg >> 6, dc = cg & 63;
            qb[(((size_t)(b*NHEAD + h)) * SEQ + s) * DHEAD + dc] = f2bf(v);
          } else if (cg < 2*DMODEL) {
            int cc = cg - DMODEL, h = cc >> 6, dc = cc & 63;
            size_t idx = (((size_t)(b*NHEAD + h)) * SEQ + s) * DHEAD + dc;
            outK[idx] = v;
            kb[idx] = f2bf(v);
          } else {
            int cc = cg - 2*DMODEL, h = cc >> 6, dc = cc & 63;
            size_t idx = (((size_t)(b*NHEAD + h)) * SEQ + s) * DHEAD + dc;
            outV[idx] = v;
            vt[((size_t)(b*NHEAD + h) * DHEAD + dc) * SEQ + s] = f2bf(v);
          }
        }
      }
    }
  }
}

// ---------------------------------------------------------------------------
// Causal flash attention, double-buffered TDM-staged K/V tiles.
// Block: 128 threads (4 waves), each wave owns a 16-row Q tile.
// grid = (B*H, S/64). Output bf16 [B,S,D].
// ---------------------------------------------------------------------------
#define KSTR 72   // K LDS row stride: 128B row + 16B TDM pad = 144B
#define VSTR 40   // V LDS row stride: 64B row + 16B pad = 80B
#define PSTR 40

__global__ __launch_bounds__(128) void attn_kernel(
    const bf16_t* __restrict__ qb, const bf16_t* __restrict__ kb,
    const bf16_t* __restrict__ vt, bf16_t* __restrict__ aout)
{
  __shared__ bf16_t ldsK[2][32 * KSTR];    // [kv][d], double buffered
  __shared__ bf16_t ldsV[2][64 * VSTR];    // [d][kv], double buffered
  __shared__ bf16_t ldsP[4 * 16 * PSTR];   // per-wave [m][kv]

  const int tid  = threadIdx.x;
  const int lane = tid & 31;
  const int wid  = tid >> 5;
  const int bh   = blockIdx.x;             // b*16 + h
  const int q0b  = blockIdx.y * 64;
  const int q0w  = q0b + wid * 16;

  const int ml    = lane & 15;
  const int rofs  = (lane < 16) ? 0 : 8;
  const int abase = (lane < 16) ? 0 : 8;
  const int kbb   = (lane < 16) ? 0 : 16;
  const unsigned ldsKa[2] = { (unsigned)(uintptr_t)&ldsK[0][0],
                              (unsigned)(uintptr_t)&ldsK[1][0] };
  const unsigned ldsVa[2] = { (unsigned)(uintptr_t)&ldsV[0][0],
                              (unsigned)(uintptr_t)&ldsV[1][0] };

  // --- Q fragments (already bf16, A-frag layout via two 16B loads each) ---
  const bf16_t* Qp = qb + ((size_t)bh * SEQ + q0w + ml) * DHEAD;
  Frag16 qf[2];
  #pragma unroll
  for (int f = 0; f < 2; ++f) {
    int dA = f*32 + abase;
    qf[f].h[0] = *(const v8bf*)(Qp + dA);
    qf[f].h[1] = *(const v8bf*)(Qp + dA + 16);
  }

  v8f O[4]; float mrun[8], lrun[8];
  #pragma unroll
  for (int dt = 0; dt < 4; ++dt) O[dt] = vzero8();
  #pragma unroll
  for (int r = 0; r < 8; ++r) { mrun[r] = -3.0e38f; lrun[r] = 0.f; }

  const bf16_t* Kbase = kb + (size_t)bh * SEQ * DHEAD;
  const bf16_t* Vbase = vt + (size_t)bh * DHEAD * SEQ;

  const int nkv = (q0b + 64) >> 5;   // uniform per block: barriers convergent
  if (wid == 0) {   // prologue: stage kv-tile 0
    tdm_load_2d(ldsKa[0], Kbase,
                (unsigned)DHEAD, (unsigned)SEQ, 64u, 32u, (unsigned)DHEAD, 4u, 3u);
    tdm_load_2d(ldsVa[0], Vbase,
                (unsigned)SEQ, (unsigned)DHEAD, 32u, 64u, (unsigned)SEQ, 3u, 3u);
  }

  for (int i = 0; i < nkv; ++i) {
    const int cur = i & 1;
    const int kv0 = i << 5;
    if (wid == 0) {
      if (i + 1 < nkv) {   // prefetch next kv tile into the other buffer
        int kv1 = (i + 1) << 5;
        tdm_load_2d(ldsKa[cur ^ 1], Kbase + (size_t)kv1 * DHEAD,
                    (unsigned)DHEAD, (unsigned)SEQ, 64u, 32u, (unsigned)DHEAD, 4u, 3u);
        tdm_load_2d(ldsVa[cur ^ 1], Vbase + (size_t)kv1,
                    (unsigned)SEQ, (unsigned)DHEAD, 32u, 64u, (unsigned)SEQ, 3u, 3u);
        __builtin_amdgcn_s_wait_tensorcnt(2);
      } else {
        __builtin_amdgcn_s_wait_tensorcnt(0);
      }
    }
    __syncthreads();

    // --- scores: S = Q K^T  (two 16x16 C tiles, dh=64 -> 2 k-steps) ---
    const bf16_t* Kt = &ldsK[cur][0];
    const bf16_t* Vt = &ldsV[cur][0];
    v8f sc[2];
    #pragma unroll
    for (int t = 0; t < 2; ++t) {
      v8f c = vzero8();
      #pragma unroll
      for (int ks = 0; ks < 2; ++ks) {
        Frag16 kf;
        const bf16_t* kr = &Kt[(t*16 + ml)*KSTR + ks*32 + abase];
        kf.h[0] = *(const v8bf*)(kr);
        kf.h[1] = *(const v8bf*)(kr + 16);
        c = __builtin_amdgcn_wmma_f32_16x16x32_bf16(
            false, qf[ks].v, false, kf.v, (short)0, c, false, false);
      }
      sc[t] = c;
    }

    // --- scale + causal mask (reference's exact -1e10) ---
    #pragma unroll
    for (int t = 0; t < 2; ++t) {
      #pragma unroll
      for (int r = 0; r < 8; ++r) {
        int rg = q0w + r + rofs;
        int cg = kv0 + t*16 + ml;
        float s = sc[t][r] * 0.125f;   // 1/sqrt(64)
        sc[t][r] = (cg <= rg) ? s : -1.0e10f;
      }
    }

    // --- online softmax; rows live across 16-lane halves per VGPR r ---
    float p0v[8], p1v[8];
    #pragma unroll
    for (int r = 0; r < 8; ++r) {
      float mc = fmaxf(sc[0][r], sc[1][r]);
      mc = fmaxf(mc, __shfl_xor(mc, 1));
      mc = fmaxf(mc, __shfl_xor(mc, 2));
      mc = fmaxf(mc, __shfl_xor(mc, 4));
      mc = fmaxf(mc, __shfl_xor(mc, 8));
      float mn = fmaxf(mrun[r], mc);
      float alpha = __expf(mrun[r] - mn);
      float p0 = __expf(sc[0][r] - mn);
      float p1 = __expf(sc[1][r] - mn);
      float ps = p0 + p1;
      ps += __shfl_xor(ps, 1);
      ps += __shfl_xor(ps, 2);
      ps += __shfl_xor(ps, 4);
      ps += __shfl_xor(ps, 8);
      lrun[r] = lrun[r] * alpha + ps;
      mrun[r] = mn;
      #pragma unroll
      for (int dt = 0; dt < 4; ++dt) O[dt][r] *= alpha;
      p0v[r] = p0; p1v[r] = p1;
    }

    // --- reshape P through per-wave LDS into A-frag layout ---
    bf16_t* pw = &ldsP[wid * 16 * PSTR];
    #pragma unroll
    for (int r = 0; r < 8; ++r) {
      int rl = r + rofs;
      pw[rl*PSTR + ml]      = f2bf(p0v[r]);
      pw[rl*PSTR + 16 + ml] = f2bf(p1v[r]);
    }
    Frag16 pf;
    pf.h[0] = *(const v8bf*)&pw[ml*PSTR + abase];
    pf.h[1] = *(const v8bf*)&pw[ml*PSTR + 16 + abase];

    // --- O += P V ---
    #pragma unroll
    for (int dt = 0; dt < 4; ++dt) {
      Frag16 vf;
      const bf16_t* vr = &Vt[(dt*16 + ml)*VSTR + abase];
      vf.h[0] = *(const v8bf*)(vr);
      vf.h[1] = *(const v8bf*)(vr + 16);
      O[dt] = __builtin_amdgcn_wmma_f32_16x16x32_bf16(
          false, pf.v, false, vf.v, (short)0, O[dt], false, false);
    }
    __syncthreads();
  }

  // --- normalize, write bf16 a in [B,S,D] for the proj GEMM ---
  const int b = bh >> 4, h = bh & 15;
  #pragma unroll
  for (int dt = 0; dt < 4; ++dt) {
    #pragma unroll
    for (int r = 0; r < 8; ++r) {
      int rg = q0w + r + rofs;
      float v = O[dt][r] / lrun[r];
      aout[((size_t)b * SEQ + rg) * DMODEL + h*DHEAD + dt*16 + ml] = f2bf(v);
    }
  }
}

// ---------------------------------------------------------------------------
// Launcher
// ---------------------------------------------------------------------------
extern "C" void kernel_launch(void* const* d_in, const int* in_sizes, int n_in,
                              void* d_out, int out_size, void* d_ws, size_t ws_size,
                              hipStream_t stream) {
  const float* x      = (const float*)d_in[0];   // [4,2048,1024]
  const float* w_attn = (const float*)d_in[1];   // [1024,3072]
  const float* b_attn = (const float*)d_in[2];   // [3072]
  const float* w_proj = (const float*)d_in[3];   // [1024,1024]
  const float* b_proj = (const float*)d_in[4];   // [1024]

  float* out  = (float*)d_out;                   // [4,2048,1024]
  float* outK = out + (size_t)QSZ;               // present[0] : [B,H,S,dh] f32
  float* outV = out + (size_t)2*QSZ;             // present[1] : [B,H,S,dh] f32

  bf16_t* ws  = (bf16_t*)d_ws;
  bf16_t* xb  = ws;                              // x   bf16   [8192,1024]
  bf16_t* waT = xb  + (size_t)QSZ;               // w_attn^T   [3072,1024]
  bf16_t* wpT = waT + (size_t)3*DMODEL*DMODEL;   // w_proj^T   [1024,1024]
  bf16_t* qb  = wpT + (size_t)DMODEL*DMODEL;     // Q bf16 [B,H,S,dh]
  bf16_t* kbf = qb  + (size_t)QSZ;               // K bf16 [B,H,S,dh]
  bf16_t* vtb = kbf + (size_t)QSZ;               // V bf16 [B,H,dh,S] (transposed)
  bf16_t* ab  = vtb + (size_t)QSZ;               // attn out bf16 [B,S,D]

  // 0) one-time precision/layout conversion
  cvt_bf16_kernel<<<QSZ/4/256, 256, 0, stream>>>(x, xb, QSZ/4);
  dim3 gt1(3*DMODEL/32, DMODEL/32);              // (96, 32)
  cvt_transpose_kernel<<<gt1, 256, 0, stream>>>(w_attn, waT, DMODEL, 3*DMODEL);
  dim3 gt2(DMODEL/32, DMODEL/32);                // (32, 32)
  cvt_transpose_kernel<<<gt2, 256, 0, stream>>>(w_proj, wpT, DMODEL, DMODEL);

  // 1) QKV = x @ w_attn + b_attn  (Q/K/V routed; K,V also to present in f32)
  dim3 g1(3*DMODEL/128, (BATCH*SEQ)/128);        // (24, 64)
  gemm_bf16_kernel<<<g1, 256, 0, stream>>>(xb, waT, b_attn,
                                           nullptr, qb, outK, kbf, outV, vtb,
                                           BATCH*SEQ, 3*DMODEL, DMODEL, 0);
  // 2) causal attention per (b,h)
  dim3 g2(BATCH*NHEAD, SEQ/64);                  // (64, 32)
  attn_kernel<<<g2, 128, 0, stream>>>(qb, kbf, vtb, ab);

  // 3) out = a @ w_proj + b_proj
  dim3 g3(DMODEL/128, (BATCH*SEQ)/128);          // (8, 64)
  gemm_bf16_kernel<<<g3, 256, 0, stream>>>(ab, wpT, b_proj,
                                           out, nullptr, nullptr, nullptr,
                                           nullptr, nullptr,
                                           BATCH*SEQ, DMODEL, DMODEL, 1);
}
